// RWKVTimeMix_4440996184661
// MI455X (gfx1250) — matrix operand order repaired and checked by reading, not compiled, not run on hardware
//
#include <hip/hip_runtime.h>
#include <hip/hip_bf16.h>
#include <stdint.h>

#define Bn 4
#define Tn 4096
#define Dn 1024
#define Mn (Bn * Tn)  // 16384

#define KSTEP 64
#define NSTAGE (Dn / KSTEP)  // 16
#define LDSS 72              // padded LDS row stride (elements) to spread banks

typedef __attribute__((ext_vector_type(16))) __bf16 v16bf;
typedef __attribute__((ext_vector_type(8)))  float  v8f;
typedef __attribute__((ext_vector_type(4)))  int    vi4;

typedef __attribute__((address_space(1))) vi4 as1_vi4;
typedef __attribute__((address_space(3))) vi4 as3_vi4;

union FragBF {
    uint4 u[2];
    v16bf v;
};

#if __has_builtin(__builtin_amdgcn_global_load_async_to_lds_b128)
#define HAS_ASYNC 1
#endif

#ifdef HAS_ASYNC
#if __has_builtin(__builtin_amdgcn_s_wait_asynccnt)
#define WAIT_ASYNC(n) __builtin_amdgcn_s_wait_asynccnt(n)
#else
#define WAIT_ASYNC(n) asm volatile("s_wait_asynccnt " #n)
#endif
#else
#define WAIT_ASYNC(n)
#endif

__device__ __forceinline__ void async_ld16(const unsigned short* g, unsigned short* l) {
#ifdef HAS_ASYNC
    __builtin_amdgcn_global_load_async_to_lds_b128(
        (as1_vi4*)(__attribute__((address_space(1))) void*)(g),
        (as3_vi4*)(__attribute__((address_space(3))) void*)(l), 0, 0);
#else
    *(uint4*)l = *(const uint4*)g;
#endif
}

__device__ __forceinline__ unsigned short f2bf(float f) {
    // round-to-nearest-even f32 -> bf16
    unsigned int x = __float_as_uint(f);
    unsigned int r = (x + 0x7fffu + ((x >> 16) & 1u)) >> 16;
    return (unsigned short)r;
}

// ---------------- weight f32 -> bf16 ----------------
__global__ void cvt_bf16_kernel(const float* __restrict__ src,
                                unsigned short* __restrict__ dst, int n) {
    int i = blockIdx.x * blockDim.x + threadIdx.x;
    if (i < n) dst[i] = f2bf(src[i]);
}

// ---------------- token-shift mix (writes bf16 xk/xv/xr) ----------------
__global__ void mix_kernel(const float* __restrict__ x,
                           const float* __restrict__ tmk,
                           const float* __restrict__ tmv,
                           const float* __restrict__ tmr,
                           unsigned short* __restrict__ xk,
                           unsigned short* __restrict__ xv,
                           unsigned short* __restrict__ xr) {
    int i = blockIdx.x * blockDim.x + threadIdx.x;  // over M*D
    int d = i & (Dn - 1);
    int t = (i >> 10) & (Tn - 1);  // i/D mod T
    float xc = x[i];
    float xp = (t == 0) ? 0.0f : x[i - Dn];
    float mk = tmk[d], mv = tmv[d], mr = tmr[d];
    xk[i] = f2bf(xc * mk + xp * (1.0f - mk));
    xv[i] = f2bf(xc * mv + xp * (1.0f - mv));
    xr[i] = f2bf(xc * mr + xp * (1.0f - mr));
}

// Stage one K-step tile (A: 128x64, B: 64x64 bf16) into LDS via async copies.
// 256 threads: A -> thread covers 64B of one row (4 x b128);
//              B -> thread covers 32B of one row (2 x b128).
__device__ __forceinline__ void prefetch_stage(const unsigned short* __restrict__ A,
                                               const unsigned short* __restrict__ W,
                                               int mBase, int nBase, int k0,
                                               unsigned short* sa, unsigned short* sb,
                                               int tid) {
    int rowA = tid >> 1;
    int segA = (tid & 1) * 32;
    const unsigned short* ga = A + (size_t)(mBase + rowA) * Dn + k0 + segA;
    unsigned short* la = sa + rowA * LDSS + segA;
#pragma unroll
    for (int i = 0; i < 4; ++i) async_ld16(ga + i * 8, la + i * 8);

    int rowB = tid >> 2;
    int segB = (tid & 3) * 16;
    const unsigned short* gb = W + (size_t)(nBase + rowB) * Dn + k0 + segB;
    unsigned short* lb = sb + rowB * LDSS + segB;
#pragma unroll
    for (int i = 0; i < 2; ++i) async_ld16(gb + i * 8, lb + i * 8);
}

// ---------------- NT GEMM: out[M,N] = A[M,K](bf16) * W[N,K](bf16)^T ----------------
// Block: 256 thr = 8 waves; block tile 128(M) x 64(N); B tile shared via LDS;
// double-buffered async global->LDS pipeline; each wave owns a 16x64 slice.
__global__ void __launch_bounds__(256)
gemm_bf16_kernel(const unsigned short* __restrict__ A,
                 const unsigned short* __restrict__ W,
                 float* __restrict__ out) {
    __shared__ unsigned short sA[2][128 * LDSS];  // 36 KB
    __shared__ unsigned short sB[2][64 * LDSS];   // 18 KB

    const int N = Dn;
    int tid   = threadIdx.x;
    int wId   = tid >> 5;
    int lane  = tid & 31;
    int h     = lane >> 4;
    int m16   = lane & 15;
    int mBase = blockIdx.x * 128;
    int nBase = blockIdx.y * 64;

    v8f c[4] = {};

    prefetch_stage(A, W, mBase, nBase, 0, sA[0], sB[0], tid);
    prefetch_stage(A, W, mBase, nBase, KSTEP, sA[1], sB[1], tid);

    int buf = 0;
    for (int ks = 0; ks < NSTAGE; ++ks) {
        if (ks + 1 < NSTAGE) {
            WAIT_ASYNC(6);   // stage ks done; stage ks+1 may still be in flight
        } else {
            WAIT_ASYNC(0);
        }
        __syncthreads();     // stage-ks tile visible to all waves

        const unsigned short* sa = sA[buf];
        const unsigned short* sb = sB[buf];
        const unsigned short* aRow = sa + (wId * 16 + m16) * LDSS;
#pragma unroll
        for (int ksub = 0; ksub < KSTEP; ksub += 32) {
            FragBF a;
            a.u[0] = *(const uint4*)(aRow + ksub + 8 * h);
            a.u[1] = *(const uint4*)(aRow + ksub + 16 + 8 * h);
#pragma unroll
            for (int nn = 0; nn < 4; ++nn) {
                const unsigned short* bRow = sb + (nn * 16 + m16) * LDSS + ksub + 8 * h;
                FragBF b;
                b.u[0] = *(const uint4*)(bRow);
                b.u[1] = *(const uint4*)(bRow + 16);
                c[nn] = __builtin_amdgcn_wmma_f32_16x16x32_bf16(
                    false, a.v, false, b.v, (short)0, c[nn], false, false);
            }
        }

        __syncthreads();     // all waves done reading buf before it is refilled
        if (ks + 2 < NSTAGE)
            prefetch_stage(A, W, mBase, nBase, (ks + 2) * KSTEP, sA[buf], sB[buf], tid);
        buf ^= 1;
    }

#pragma unroll
    for (int nn = 0; nn < 4; ++nn) {
        int col = nBase + nn * 16 + m16;
#pragma unroll
        for (int r = 0; r < 8; ++r) {
            size_t row = (size_t)(mBase + wId * 16 + r + 8 * h);
            out[row * N + col] = c[nn][r];
        }
    }
}

// ---------------- WKV scan (sequential in T), fused sigmoid(r)*wkv -> bf16 y ----------------
__global__ void wkv_scan_kernel(const float* __restrict__ kin,
                                const float* __restrict__ vin,
                                const float* __restrict__ rin,
                                const float* __restrict__ wdec,
                                const float* __restrict__ ufirst,
                                unsigned short* __restrict__ y) {
    int i = blockIdx.x * blockDim.x + threadIdx.x;  // 0 .. B*D-1
    int d = i & (Dn - 1);
    int b = i >> 10;
    float w = wdec[d];
    float u = ufirst[d];
    float p = -3.0e38f;  // ~ -inf
    float q = 0.0f;
    size_t base = (size_t)b * Tn * Dn + d;
    for (int t = 0; t < Tn; ++t) {
        size_t idx = base + (size_t)t * Dn;
        float kt = kin[idx], vt = vin[idx], rt = rin[idx];
        float uk  = u + kt;
        float np  = fmaxf(p, uk);
        float so  = __expf(fmaxf(p - np, -30.0f));
        float sc  = __expf(fmaxf(uk - np, -30.0f));
        float wkv = (so * q + sc * vt) / (so + sc + 1e-9f);
        float pw  = p + w;
        float np2 = fmaxf(pw, kt);
        float sp  = __expf(fmaxf(pw - np2, -30.0f));
        float sn  = __expf(fmaxf(kt - np2, -30.0f));
        q = sp * q + sn * vt;
        p = np2;
        float sig = 1.0f / (1.0f + __expf(-rt));
        y[idx] = f2bf(sig * wkv);
    }
}

extern "C" void kernel_launch(void* const* d_in, const int* in_sizes, int n_in,
                              void* d_out, int out_size, void* d_ws, size_t ws_size,
                              hipStream_t stream) {
    const float* x   = (const float*)d_in[0];
    const float* tmk = (const float*)d_in[1];
    const float* tmv = (const float*)d_in[2];
    const float* tmr = (const float*)d_in[3];
    const float* Wk  = (const float*)d_in[4];
    const float* Wv  = (const float*)d_in[5];
    const float* Wr  = (const float*)d_in[6];
    const float* Wo  = (const float*)d_in[7];
    const float* td  = (const float*)d_in[8];
    const float* tf  = (const float*)d_in[9];

    char* ws = (char*)d_ws;
    const size_t MB = 1ull << 20;
    unsigned short* xk   = (unsigned short*)(ws);             // 32 MB
    unsigned short* xv   = (unsigned short*)(ws + 32 * MB);   // 32 MB
    unsigned short* xr   = (unsigned short*)(ws + 64 * MB);   // 32 MB
    float*          kbuf = (float*)(ws + 96 * MB);            // 64 MB
    float*          vbuf = (float*)(ws + 160 * MB);           // 64 MB
    float*          rbuf = (float*)(ws + 224 * MB);           // 64 MB
    unsigned short* wkb  = (unsigned short*)(ws + 288 * MB);  // 2 MB each
    unsigned short* wvb  = (unsigned short*)(ws + 290 * MB);
    unsigned short* wrb  = (unsigned short*)(ws + 292 * MB);
    unsigned short* wob  = (unsigned short*)(ws + 294 * MB);
    unsigned short* ybuf = xk;  // xk dead after its GEMM -> reuse for y (bf16)

    const int nW = Dn * Dn;  // 1M elements per weight
    cvt_bf16_kernel<<<nW / 256, 256, 0, stream>>>(Wk, wkb, nW);
    cvt_bf16_kernel<<<nW / 256, 256, 0, stream>>>(Wv, wvb, nW);
    cvt_bf16_kernel<<<nW / 256, 256, 0, stream>>>(Wr, wrb, nW);
    cvt_bf16_kernel<<<nW / 256, 256, 0, stream>>>(Wo, wob, nW);

    mix_kernel<<<(Mn * Dn) / 256, 256, 0, stream>>>(x, tmk, tmv, tmr, xk, xv, xr);

    dim3 gemmGrid(Mn / 128, Dn / 64);  // (128, 16)
    gemm_bf16_kernel<<<gemmGrid, 256, 0, stream>>>(xk, wkb, kbuf);
    gemm_bf16_kernel<<<gemmGrid, 256, 0, stream>>>(xv, wvb, vbuf);
    gemm_bf16_kernel<<<gemmGrid, 256, 0, stream>>>(xr, wrb, rbuf);

    wkv_scan_kernel<<<(Bn * Dn) / 256, 256, 0, stream>>>(kbuf, vbuf, rbuf, td, tf, ybuf);

    gemm_bf16_kernel<<<gemmGrid, 256, 0, stream>>>(ybuf, wob, (float*)d_out);
}
